// TLVAE_53764400611510
// MI455X (gfx1250) — matrix operand order repaired and checked by reading, not compiled, not run on hardware
//
#include <hip/hip_runtime.h>
#include <hip/hip_bf16.h>
#include <math.h>

typedef __attribute__((ext_vector_type(16))) _Float16 v16h;
typedef __attribute__((ext_vector_type(8)))  _Float16 v8h;
typedef __attribute__((ext_vector_type(2)))  _Float16 v2h;
typedef __attribute__((ext_vector_type(8)))  float    v8f;

#define BB   8192
#define DXX  192
#define WW   64
#define TT   128
#define DZZ  16
#define DHH  32
#define DGG  64

// ---------------------------------------------------------------------------
// Generic WMMA GEMM:  out[M=8192, N] = epilogue( A @ Wt + bias )
// A gathered from up to 3 row-major f32 segments (concat along K, widths even).
// Wt is pre-converted f16, layout [KSTEPS*32 x N], zero padded beyond K.
// MODE 0: store raw, 1: relu+store, 2: sigmoid + fused MSE atomic (no store)
// ---------------------------------------------------------------------------
template<int KSTEPS, int MODE>
__global__ __launch_bounds__(128) void gemm_wmma_k(
    const float* a0, int w0, int s0,
    const float* a1, int w1, int s1,
    const float* a2, int w2, int s2,
    int K, const _Float16* Wt,
    const float* bias, float* out, int N,
    const float* cmp, int cmpStride, float* acc)
{
    __shared__ _Float16 sA[64 * 32];    // A tile, row-major [row][k]
    __shared__ _Float16 sBt[16 * 32];   // B tile transposed [n][k]
    const int tid  = threadIdx.x;
    const int lane = tid & 31;
    const int wave = tid >> 5;
    const int mBlk = blockIdx.x * 64;
    const int n0   = blockIdx.y * 16;

    const int half = lane >> 4;
    const int mr   = lane & 15;
    const _Float16* rowA = &sA[(wave * 16 + mr) * 32];
    const _Float16* rowB = &sBt[mr * 32];

    // A staging: thread owns an even k pair, streams rows (segment pick hoisted)
    const int kk2 = (tid & 15) * 2;
    const int rb0 = tid >> 4;

    v8f c = {};
#pragma unroll
    for (int ks = 0; ks < KSTEPS; ++ks) {
        const int k0 = ks << 5;
        // ---- stage A tile (64 rows x 32 k) as packed f16 pairs ----
        {
            int gk = k0 + kk2;
            const float* src = nullptr;
            int stride = 0;
            if (gk < K) {
                if (gk < w0)                { src = a0 + gk;             stride = s0; }
                else if (gk - w0 < w1)      { src = a1 + (gk - w0);      stride = s1; }
                else if (gk - w0 - w1 < w2) { src = a2 + (gk - w0 - w1); stride = s2; }
            }
#pragma unroll
            for (int r = rb0; r < 64; r += 8) {
                float v0 = 0.0f, v1 = 0.0f;
                if (src) {
                    const float* p = src + (size_t)(mBlk + r) * stride;
                    v0 = p[0]; v1 = p[1];
                }
                v2h pr; pr[0] = (_Float16)v0; pr[1] = (_Float16)v1;
                *(v2h*)&sA[r * 32 + kk2] = pr;
            }
        }
        // ---- stage B tile transposed (16 n x 32 k) as packed pairs ----
#pragma unroll
        for (int p = tid; p < 256; p += 128) {
            int n  = p >> 4;
            int k2 = (p & 15) * 2;
            v2h pr;
            pr[0] = Wt[(size_t)(k0 + k2)     * N + n0 + n];
            pr[1] = Wt[(size_t)(k0 + k2 + 1) * N + n0 + n];
            *(v2h*)&sBt[n * 32 + k2] = pr;
        }
        __syncthreads();

        // ---- fragments via contiguous b128 LDS loads ----
        // A 16-bit layout: lane half=0 -> K [0..7]+[16..23], half=1 -> [8..15]+[24..31]
        v8h a_lo = *(const v8h*)(rowA + 8 * half);
        v8h a_hi = *(const v8h*)(rowA + 16 + 8 * half);
        // B 16-bit layout: N = lane%16, half=0 -> K [0..7]+[16..23], half=1 -> [8..15]+[24..31]
        v8h b_lo = *(const v8h*)(rowB + 8 * half);
        v8h b_hi = *(const v8h*)(rowB + 16 + 8 * half);
        v16h af = __builtin_shufflevector(a_lo, a_hi, 0,1,2,3,4,5,6,7,8,9,10,11,12,13,14,15);
        v16h bf = __builtin_shufflevector(b_lo, b_hi, 0,1,2,3,4,5,6,7,8,9,10,11,12,13,14,15);
        c = __builtin_amdgcn_wmma_f32_16x16x32_f16(false, af, false, bf,
                                                   (short)0, c, false, false);
        __syncthreads();
    }

    // epilogue: C/D layout: N = lane%16, M = r + 8*(lane/16)
    const int gn = n0 + mr;
    const float bv = bias[gn];
    float lsum = 0.0f;
#pragma unroll
    for (int r = 0; r < 8; ++r) {
        int gm = mBlk + wave * 16 + half * 8 + r;
        float v = c[r] + bv;
        if (MODE == 1) v = fmaxf(v, 0.0f);
        if (MODE == 2) {
            float sg = 1.0f / (1.0f + expf(-v));
            float d  = cmp[(size_t)gm * cmpStride + gn] - sg;
            lsum += d * d;
        } else {
            out[(size_t)gm * N + gn] = v;
        }
    }
    if (MODE == 2) atomicAdd(acc, lsum);
}

// ---------------------------------------------------------------------------
// BatchNorm statistics (training mode, biased variance): one block per column
// stats[c] = mean, stats[32+c] = rsqrt(var + 1e-5)
// ---------------------------------------------------------------------------
__global__ __launch_bounds__(256) void bn_stats_k(const float* pre, float* stats)
{
    __shared__ float s1[256], s2[256];
    const int c = blockIdx.x;
    float a = 0.0f, b = 0.0f;
    for (int r = threadIdx.x; r < BB; r += 256) {
        float v = pre[(size_t)r * 32 + c];
        a += v; b += v * v;
    }
    s1[threadIdx.x] = a; s2[threadIdx.x] = b;
    __syncthreads();
    for (int o = 128; o > 0; o >>= 1) {
        if (threadIdx.x < o) { s1[threadIdx.x] += s1[threadIdx.x + o];
                               s2[threadIdx.x] += s2[threadIdx.x + o]; }
        __syncthreads();
    }
    if (threadIdx.x == 0) {
        float m = s1[0] * (1.0f / BB);
        float v = s2[0] * (1.0f / BB) - m * m;
        stats[c]      = m;
        stats[32 + c] = rsqrtf(v + 1e-5f);
    }
}

__device__ __forceinline__ float softplus_f(float x) {
    return fmaxf(x, 0.0f) + log1pf(expf(-fabsf(x)));
}
__device__ __forceinline__ float kld_el(float m1, float s1, float m2, float s2) {
    const float EPS = 1e-10f;
    float a = s1 + EPS, b = s2 + EPS;
    return 2.0f * (logf(b) - logf(a)) + (a * a + (m1 - m2) * (m1 - m2)) / (b * b) - 1.0f;
}

// ---------------------------------------------------------------------------
// One ladder level: BN+LeakyReLU both blocks, mu/var heads, reparam, KL accum
// ---------------------------------------------------------------------------
__global__ __launch_bounds__(256) void apply_level_k(
    const float* preE, const float* stE, const float* gE, const float* beE,
    const float* WmE, const float* bmE, const float* WvE, const float* bvE,
    const float* preD, const float* stD, const float* gD, const float* beD,
    const float* WmD, const float* bmD, const float* WvD, const float* bvD,
    const float* ej, float* z, float* kl, int level0)
{
    __shared__ float sWmE[512], sWvE[512], sWmD[512], sWvD[512];
    for (int i = threadIdx.x; i < 512; i += 256) {
        sWmE[i] = WmE[i]; sWvE[i] = WvE[i]; sWmD[i] = WmD[i]; sWvD[i] = WvD[i];
    }
    __syncthreads();

    const int b = blockIdx.x * 256 + threadIdx.x;
    float hE[32], hD[32];
#pragma unroll
    for (int c = 0; c < 32; ++c) {
        float vE = (preE[(size_t)b * 32 + c] - stE[c]) * stE[32 + c] * gE[c] + beE[c];
        hE[c] = (vE > 0.0f) ? vE : 0.01f * vE;     // encoder slope
        float vD = (preD[(size_t)b * 32 + c] - stD[c]) * stD[32 + c] * gD[c] + beD[c];
        hD[c] = (vD > 0.0f) ? vD : 0.1f * vD;      // decoder/prior slope
    }
    float klsum = 0.0f;
#pragma unroll
    for (int o = 0; o < 16; ++o) {
        float me = bmE[o], ve = bvE[o], mp = bmD[o], vp = bvD[o];
#pragma unroll
        for (int c = 0; c < 32; ++c) {
            me += hE[c] * sWmE[c * 16 + o];
            ve += hE[c] * sWvE[c * 16 + o];
            mp += hD[c] * sWmD[c * 16 + o];
            vp += hD[c] * sWvD[c * 16 + o];
        }
        ve = softplus_f(ve);
        vp = softplus_f(vp);
        float e = ej[(size_t)b * DZZ + o];
        if (level0) {
            z[(size_t)b * DZZ + o] = me + expf(ve) * e;   // std = exp(delta)
            klsum += kld_el(me, ve, mp, vp);
        } else {
            z[(size_t)b * DZZ + o] = me + sqrtf(ve) * e;  // std = sqrt(var)
            klsum += kld_el(mp, vp, me, ve);              // swapped order
        }
    }
    kl[b] += klsum;
}

// GRU gate update (biases already folded into gi/gh by GEMM epilogue)
__global__ __launch_bounds__(256) void gru_gate_k(const float* gi, const float* gh, float* h)
{
    const int t = blockIdx.x * 256 + threadIdx.x;
    const int b = t >> 6, k = t & 63;
    const size_t o = (size_t)b * 192;
    float r = 1.0f / (1.0f + expf(-(gi[o + k]       + gh[o + k])));
    float u = 1.0f / (1.0f + expf(-(gi[o + 64 + k]  + gh[o + 64 + k])));
    float n = tanhf(gi[o + 128 + k] + r * gh[o + 128 + k]);
    h[t] = (1.0f - u) * n + u * h[t];
}

__global__ __launch_bounds__(256) void init_k(const float* eps0, float* z, float* h,
                                              float* kl, float* acc)
{
    const int t = blockIdx.x * 256 + threadIdx.x;
    if (t < BB * DGG) h[t] = 0.0f;
    if (t < BB * DZZ) z[t] = eps0[t] * expf(1.0f);
    if (t < BB)       kl[t] = 0.0f;
    if (t == 0) { acc[0] = 0.0f; acc[1] = 0.0f; }
}

__global__ __launch_bounds__(256) void cvt_w_k(const float* src, _Float16* dst,
                                               int K, int N, int Kpad, int trans)
{
    const int t = blockIdx.x * 256 + threadIdx.x;
    if (t >= Kpad * N) return;
    const int kp = t / N, n = t % N;
    float v = 0.0f;
    if (kp < K) v = trans ? src[(size_t)n * K + kp] : src[(size_t)kp * N + n];
    dst[t] = (_Float16)v;
}

__global__ __launch_bounds__(256) void final_k(const float* z, const float* h,
                                               const float* kl, const float* Wp,
                                               const float* bp, const float* y,
                                               float* out, float* acc)
{
    const int b = blockIdx.x * 256 + threadIdx.x;
    if (b >= BB) return;
    out[b] = kl[b];
    float s = bp[0];
#pragma unroll
    for (int i = 0; i < 16; ++i) s += z[(size_t)b * 16 + i] * Wp[i];
#pragma unroll
    for (int i = 0; i < 64; ++i) s += h[(size_t)b * 64 + i] * Wp[16 + i];
    float xp = 1.0f / (1.0f + expf(-s));
    out[BB + 2 + b] = xp;
    float d = xp - y[b];
    atomicAdd(&acc[1], d * d);
}

__global__ void final2_k(const float* acc, float* out)
{
    out[BB]     = acc[0] * (1.0f / (float)(BB * WW));  // rec (sum of means)
    out[BB + 1] = acc[1] * (1.0f / (float)BB);         // pred_loss
}

// ---------------------------------------------------------------------------
extern "C" void kernel_launch(void* const* d_in, const int* in_sizes, int n_in,
                              void* d_out, int out_size, void* d_ws, size_t ws_size,
                              hipStream_t stream)
{
    const float* P[64];
    for (int i = 0; i < n_in && i < 64; ++i) P[i] = (const float*)d_in[i];
    const float* x    = P[0];
    const float* y    = P[1];
    const float* eps0 = P[2];
    const float* eseq = P[3];
    // params: enc_xz 4..11, dec_zxz 12..19, enc_zz 20..27, dec_zz 28..35,
    //         dec_zx 36..39, gru 40..43, pred 44..45

    float* Wf = (float*)d_ws;
    float* z    = Wf;                       // 131072
    float* h    = Wf + 131072;              // 524288
    float* kl   = Wf + 655360;              // 8192
    float* acc  = Wf + 663552;              // 2 (+2 pad)
    float* preE = Wf + 663556;              // 262144
    float* preD = Wf + 925700;              // 262144
    float* stE  = Wf + 1187844;             // 64
    float* stD  = Wf + 1187908;             // 64
    float* gi   = Wf + 1187972;             // 1572864
    float* gh   = Wf + 2760836;             // 1572864
    float* hr   = Wf + 4333700;             // 262144
    _Float16* wb = (_Float16*)(Wf + 4595844);
    _Float16* wEnc  = wb;                   // 160*32
    _Float16* wDzxz = wb + 5120;            // 96*32
    _Float16* wEzz  = wb + 8192;            // 32*32
    _Float16* wDzz  = wb + 9216;            // 96*32
    _Float16* wZx1  = wb + 12288;           // 96*32
    _Float16* wZx2  = wb + 15360;           // 32*64
    _Float16* wWiT  = wb + 17408;           // 64*192
    _Float16* wWhT  = wb + 29696;           // 64*192

    auto cvt = [&](const float* src, _Float16* dst, int K, int N, int Kpad, int tr) {
        int n = Kpad * N;
        cvt_w_k<<<(n + 255) / 256, 256, 0, stream>>>(src, dst, K, N, Kpad, tr);
    };
    cvt(P[4],  wEnc, 144, 32, 160, 0);
    cvt(P[12], wDzxz, 80, 32,  96, 0);
    cvt(P[20], wEzz,  16, 32,  32, 0);
    cvt(P[28], wDzz,  80, 32,  96, 0);
    cvt(P[36], wZx1,  80, 32,  96, 0);
    cvt(P[38], wZx2,  32, 64,  32, 0);
    cvt(P[40], wWiT,  64, 192, 64, 1);   // Wi.T
    cvt(P[41], wWhT,  64, 192, 64, 1);   // Wh.T

    init_k<<<(BB * DGG) / 256, 256, 0, stream>>>(eps0, z, h, kl, acc);

    const dim3 blk(128);
    for (int t = 0; t < TT; ++t) {
        const float* xt = x + t;  // [B,64] view, row stride 192

        // level 0: enc_xz([xt,z,h])  (K=144, Kpad=160 -> 5 ksteps)
        gemm_wmma_k<5,0><<<dim3(BB/64, 2), blk, 0, stream>>>(
            xt, 64, DXX, z, 16, 16, h, 64, 64, 144, wEnc, P[5], preE, 32,
            nullptr, 0, nullptr);
        // dec_zxz([h,z]) (K=80, Kpad=96 -> 3)
        gemm_wmma_k<3,0><<<dim3(BB/64, 2), blk, 0, stream>>>(
            h, 64, 64, z, 16, 16, nullptr, 0, 0, 80, wDzxz, P[13], preD, 32,
            nullptr, 0, nullptr);
        bn_stats_k<<<32, 256, 0, stream>>>(preE, stE);
        bn_stats_k<<<32, 256, 0, stream>>>(preD, stD);
        apply_level_k<<<BB / 256, 256, 0, stream>>>(
            preE, stE, P[6], P[7], P[8], P[9], P[10], P[11],
            preD, stD, P[14], P[15], P[16], P[17], P[18], P[19],
            eseq + (size_t)(t * 3 + 0) * BB * DZZ, z, kl, 1);

        for (int j = 1; j < 3; ++j) {
            // enc_zz(z) (K=16, Kpad=32 -> 1)
            gemm_wmma_k<1,0><<<dim3(BB/64, 2), blk, 0, stream>>>(
                z, 16, 16, nullptr, 0, 0, nullptr, 0, 0, 16, wEzz, P[21], preE, 32,
                nullptr, 0, nullptr);
            // dec_zz([z,xt]) (K=80 -> 3)
            gemm_wmma_k<3,0><<<dim3(BB/64, 2), blk, 0, stream>>>(
                z, 16, 16, xt, 64, DXX, nullptr, 0, 0, 80, wDzz, P[29], preD, 32,
                nullptr, 0, nullptr);
            bn_stats_k<<<32, 256, 0, stream>>>(preE, stE);
            bn_stats_k<<<32, 256, 0, stream>>>(preD, stD);
            apply_level_k<<<BB / 256, 256, 0, stream>>>(
                preE, stE, P[22], P[23], P[24], P[25], P[26], P[27],
                preD, stD, P[30], P[31], P[32], P[33], P[34], P[35],
                eseq + (size_t)(t * 3 + j) * BB * DZZ, z, kl, 0);
        }

        // reconstruction: hr = relu([z,h]@W1+b1) (K=80 -> 3)
        gemm_wmma_k<3,1><<<dim3(BB/64, 2), blk, 0, stream>>>(
            z, 16, 16, h, 64, 64, nullptr, 0, 0, 80, wZx1, P[37], hr, 32,
            nullptr, 0, nullptr);
        // rec += mean((xt - sigmoid(hr@W2+b2))^2) (K=32 -> 1, fused MSE)
        gemm_wmma_k<1,2><<<dim3(BB/64, 4), blk, 0, stream>>>(
            hr, 32, 32, nullptr, 0, 0, nullptr, 0, 0, 32, wZx2, P[39], hr, 64,
            xt, DXX, &acc[0]);

        // GRU pre-gates (K=64 -> 2) then gate update
        gemm_wmma_k<2,0><<<dim3(BB/64, 12), blk, 0, stream>>>(
            xt, 64, DXX, nullptr, 0, 0, nullptr, 0, 0, 64, wWiT, P[42], gi, 192,
            nullptr, 0, nullptr);
        gemm_wmma_k<2,0><<<dim3(BB/64, 12), blk, 0, stream>>>(
            h, 64, 64, nullptr, 0, 0, nullptr, 0, 0, 64, wWhT, P[43], gh, 192,
            nullptr, 0, nullptr);
        gru_gate_k<<<(BB * DGG) / 256, 256, 0, stream>>>(gi, gh, h);
    }

    final_k<<<BB / 256, 256, 0, stream>>>(z, h, kl, P[44], P[45], y, (float*)d_out, acc);
    final2_k<<<1, 1, 0, stream>>>(acc, (float*)d_out);
}
